// RNN_forecasting_26448408609202
// MI455X (gfx1250) — matrix operand order repaired and checked by reading, not compiled, not run on hardware
//
#include <hip/hip_runtime.h>

typedef __bf16 bf16;
typedef bf16  v16bf __attribute__((ext_vector_type(16)));
typedef bf16  v8bf  __attribute__((ext_vector_type(8)));
typedef bf16  v4bf  __attribute__((ext_vector_type(4)));
typedef float v8f   __attribute__((ext_vector_type(8)));

#define DLAT     512
#define BATCH    2048
#define TSTEPS   128
#define NFORE    64
#define KTILES   16      // 512 / 32 (K per bf16 WMMA)
#define NTILES   32      // 512 / 16
#define HSTRIDE  520     // padded bf16 row stride (260 dwords, %64 == 4 -> low bank conflicts)
#define MROWS    32      // batch rows per workgroup (2 WMMA M-tiles)
#define NWAVES   4       // waves per workgroup, N-dimension split
#define NTPW     (NTILES / NWAVES)   // n-tiles per wave = 8

// --- Prep: convert W_h (f32 [N=512][K=512], B(k,n)=W_h[n][k]) into bf16,
// pre-swizzled into WMMA B-fragment order:
//   wswz[ ((nt*16 + kt)*32 + lane)*16 + e ]
//   lane<16 : N = nt*16+lane,    K = kt*32 + e
//   lane>=16: N = nt*16+lane-16, K = kt*32 + 16 + e
__global__ __launch_bounds__(256) void swizzle_wh(const float* __restrict__ Wh,
                                                  bf16* __restrict__ wswz) {
    int idx  = blockIdx.x * 256 + threadIdx.x;   // 0 .. 512*512-1
    int e    = idx & 15;
    int lane = (idx >> 4) & 31;
    int kt   = (idx >> 9) & 15;
    int nt   = idx >> 13;
    int N = nt * 16 + (lane & 15);
    int K = kt * 32 + ((lane >> 4) << 4) + e;
    wswz[idx] = (bf16)Wh[N * DLAT + K];
}

__global__ __launch_bounds__(NWAVES * 32) void rnn_forecast(
    const float* __restrict__ x,     const float* __restrict__ h0,
    const float* __restrict__ W_in,  const float* __restrict__ b_h,
    const float* __restrict__ W_dec, const float* __restrict__ b_dec,
    const bf16*  __restrict__ wswz,  float* __restrict__ out) {
    __shared__ bf16  hl[MROWS * HSTRIDE];      // shared h tile (32 x 512 bf16, padded)
    __shared__ float fbuf[NWAVES * MROWS];     // decode-head partial sums

    const int tid  = threadIdx.x;
    const int lane = tid & 31;
    const int wave = tid >> 5;
    const int base = blockIdx.x * MROWS;       // this workgroup's batch rows

    // ---- load initial h (f32 -> bf16) into LDS, all 4 waves cooperate ----
    for (int i = tid; i < (MROWS * DLAT) / 4; i += NWAVES * 32) {
        int j = i * 4;
        int m = j >> 9;       // /512
        int d = j & 511;
        const float4 v = *(const float4*)(h0 + (size_t)(base + m) * DLAT + d);
        v4bf b;
        b[0] = (bf16)v.x; b[1] = (bf16)v.y; b[2] = (bf16)v.z; b[3] = (bf16)v.w;
        *(v4bf*)(hl + m * HSTRIDE + d) = b;
    }
    __syncthreads();

    // ---- persistent rotating B-fragment pipeline (4 slots, 3 tiles in flight) ----
    // W_h fragments are step-invariant, so the pipeline stays primed across steps:
    // the last n-tile of each step prefetches the first tiles for the next step.
    const bf16* const bp0 = wswz + (wave * NTPW) * (KTILES * 512) + lane * 16;
    v16bf Bq[4];
    Bq[0] = *(const v16bf*)(bp0);
    Bq[1] = *(const v16bf*)(bp0 + 512);
    Bq[2] = *(const v16bf*)(bp0 + 1024);

    // ---- one recurrence step: h = relu(xv*W_in + h @ W_h^T + b_h) ----
    // xv must hold (on every wave) the step input for batch row `lane`.
    auto do_step = [&](float xv) {
        // Hoisted input broadcast: same 16 values for every n-tile.
        float xm0[8], xm1[8];
#pragma unroll
        for (int r = 0; r < 8; ++r) {
            const int mr = (lane < 16) ? r : (8 + r);
            xm0[r] = __shfl(xv, mr, 32);
            xm1[r] = __shfl(xv, 16 + mr, 32);
        }

        // Phase 1: every wave loads all A fragments (full K) from the shared h tile.
        // A 16x32 layout (contiguous, matches swizzled B): lane<16 -> M=lane, K in {0..15}
        v16bf A[2][KTILES];
        const int arow = lane & 15;
        const int koff = (lane < 16) ? 0 : 16;
#pragma unroll
        for (int g = 0; g < 2; ++g) {
#pragma unroll
            for (int kt = 0; kt < KTILES; ++kt) {
                const bf16* p = hl + (g * 16 + arow) * HSTRIDE + kt * 32 + koff;
                v8bf lo = *(const v8bf*)(p);        // ds_load_b128
                v8bf hi = *(const v8bf*)(p + 8);    // ds_load_b128
                A[g][kt] = __builtin_shufflevector(lo, hi,
                    0, 1, 2, 3, 4, 5, 6, 7, 8, 9, 10, 11, 12, 13, 14, 15);
            }
        }
        __syncthreads();   // all reads of hl complete before anyone overwrites it

        // Phase 2: wave w computes n-tiles [8w, 8w+8) and writes its columns.
        const bf16* bp = bp0;
        for (int nti = 0; nti < NTPW; ++nti) {
            // next tile base; last iteration wraps to the wave's first tile,
            // pre-seeding the pipeline for the NEXT time step.
            const bf16* nxt = (nti + 1 < NTPW) ? (bp + KTILES * 512) : bp0;
            v8f acc0 = {0.f, 0.f, 0.f, 0.f, 0.f, 0.f, 0.f, 0.f};
            v8f acc1 = {0.f, 0.f, 0.f, 0.f, 0.f, 0.f, 0.f, 0.f};
#pragma unroll
            for (int kt = 0; kt < KTILES; ++kt) {
                const int pf = kt + 3;   // prefetch distance: 3 k-tiles
                if (pf < KTILES) {
                    Bq[pf & 3] = *(const v16bf*)(bp + pf * 512);
                } else {
                    Bq[pf & 3] = *(const v16bf*)(nxt + (pf - KTILES) * 512);
                }
                acc0 = __builtin_amdgcn_wmma_f32_16x16x32_bf16(
                    false, A[0][kt], false, Bq[kt & 3], (short)0, acc0, false, false);
                acc1 = __builtin_amdgcn_wmma_f32_16x16x32_bf16(
                    false, A[1][kt], false, Bq[kt & 3], (short)0, acc1, false, false);
            }
            // Epilogue: C layout lane<16 -> M=r, N=lane ; lane>=16 -> M=8+r, N=lane-16
            const int N   = (wave * NTPW + nti) * 16 + (lane & 15);
            const float win = W_in[N];
            const float bh  = b_h[N];
#pragma unroll
            for (int r = 0; r < 8; ++r) {
                const int mr = (lane < 16) ? r : (8 + r);
                {   // M-tile 0 : rows 0..15
                    float v = acc0[r] + xm0[r] * win + bh;
                    v = v > 0.f ? v : 0.f;
                    hl[mr * HSTRIDE + N] = (bf16)v;
                }
                {   // M-tile 1 : rows 16..31
                    float v = acc1[r] + xm1[r] * win + bh;
                    v = v > 0.f ? v : 0.f;
                    hl[(16 + mr) * HSTRIDE + N] = (bf16)v;
                }
            }
            bp = nxt;
        }
        __syncthreads();   // all writes of h_new visible before next step reads
    };

    // ---- decode head: f[row] = h[row] . W_dec + b_dec, split 4 ways over N ----
    auto dec_head = [&]() -> float {
        float p = 0.f;
#pragma unroll 4
        for (int i = 0; i < DLAT / (8 * NWAVES); ++i) {       // 16 iters of 8
            const int N = wave * (DLAT / NWAVES) + i * 8;
            v8bf hh = *(const v8bf*)(hl + lane * HSTRIDE + N);
            const float4 w0 = *(const float4*)(W_dec + N);
            const float4 w1 = *(const float4*)(W_dec + N + 4);
            p += (float)hh[0] * w0.x + (float)hh[1] * w0.y +
                 (float)hh[2] * w0.z + (float)hh[3] * w0.w +
                 (float)hh[4] * w1.x + (float)hh[5] * w1.y +
                 (float)hh[6] * w1.z + (float)hh[7] * w1.w;
        }
        fbuf[wave * MROWS + lane] = p;
        __syncthreads();
        return fbuf[lane] + fbuf[MROWS + lane] +
               fbuf[2 * MROWS + lane] + fbuf[3 * MROWS + lane] + b_dec[0];
    };

    // ---- encoder: 128 steps driven by x[t] ----
    for (int t = 0; t < TSTEPS; ++t) {
        const float xv = x[(size_t)t * BATCH + base + lane];
        do_step(xv);
    }

    // ---- decoder: f0 then 63 autoregressive steps ----
    float f = dec_head();
    if (tid < MROWS) out[base + tid] = f;
    for (int s = 1; s < NFORE; ++s) {
        do_step(f);
        f = dec_head();
        if (tid < MROWS) out[(size_t)s * BATCH + base + tid] = f;
    }
}

extern "C" void kernel_launch(void* const* d_in, const int* in_sizes, int n_in,
                              void* d_out, int out_size, void* d_ws, size_t ws_size,
                              hipStream_t stream) {
    const float* x     = (const float*)d_in[0];
    const float* h0    = (const float*)d_in[1];
    const float* W_in  = (const float*)d_in[2];
    const float* W_h   = (const float*)d_in[3];
    const float* b_h   = (const float*)d_in[4];
    const float* W_dec = (const float*)d_in[5];
    const float* b_dec = (const float*)d_in[6];
    bf16* wswz = (bf16*)d_ws;   // 512*512 bf16 = 512 KB scratch

    swizzle_wh<<<(DLAT * DLAT) / 256, 256, 0, stream>>>(W_h, wswz);
    rnn_forecast<<<BATCH / MROWS, NWAVES * 32, 0, stream>>>(
        x, h0, W_in, b_h, W_dec, b_dec, wswz, (float*)d_out);
}
